// DeformConv_2164663517663
// MI455X (gfx1250) — matrix operand order repaired and checked by reading, not compiled, not run on hardware
//
#include <hip/hip_runtime.h>

typedef __attribute__((ext_vector_type(16))) _Float16 v16h;
typedef __attribute__((ext_vector_type(8)))  float    v8f;

#define BDIM 8
#define CDIM 64
#define ODIM 64
#define HDIM 128
#define WDIM 128

// LDS layout for the fused kernel
#define AOFF_BYTES (18 * 2 * 32 * 16 * 2)          // 36864 B: offset-conv A-fragments
#define ADEF_BYTES (18 * 4 * 32 * 16 * 2)          // 73728 B: deform-conv A-fragments
#define OFFS_BYTES (8 * 16 * 18 * 4)               //  9216 B: per-wave pixel offsets
#define SMEM_BYTES (AOFF_BYTES + ADEF_BYTES + OFFS_BYTES)   // 119808 B (<160KB; 2 blocks/WGP)

// ---------------------------------------------------------------------------
// Kernel T: NCHW -> NHWC transpose of x via LDS tile (coalesced both sides).
// ---------------------------------------------------------------------------
__global__ void __launch_bounds__(256)
transpose_nchw_nhwc(const float* __restrict__ x, float* __restrict__ xn) {
  __shared__ float tile[64][65];
  int blk = blockIdx.x;            // B*H*(W/64) = 2048
  int wchunk = blk & 1;
  int bh = blk >> 1;
  int h = bh & (HDIM - 1);
  int b = bh >> 7;
  int w0 = wchunk * 64;
  int t = threadIdx.x;
#pragma unroll
  for (int i = 0; i < 16; ++i) {
    int e = t + i * 256;
    int c = e >> 6, ww = e & 63;
    tile[c][ww] = x[(((size_t)b * CDIM + c) * HDIM + h) * WDIM + w0 + ww];
  }
  __syncthreads();
#pragma unroll
  for (int i = 0; i < 16; ++i) {
    int e = t + i * 256;
    int ww = e >> 6, c = e & 63;
    xn[(((size_t)(b * HDIM + h) * WDIM) + w0 + ww) * CDIM + c] = tile[c][ww];
  }
}

// ---------------------------------------------------------------------------
// A-fragment packing, contiguous k order (K_rel = 16*(lane/16) + h).
// Reduction index = k*64 + c, 18 chunks of 32.
//   lane<16:  h=0..15 -> K_rel 0..15   (M = lane)
//   lane>=16: h=0..15 -> K_rel 16..31   (M = lane-16)
// ---------------------------------------------------------------------------

// Offset conv weights: (18, 64, 3, 3); M padded 18 -> 32 (2 mtiles).
__global__ void __launch_bounds__(256)
pack_woffset(const float* __restrict__ wo, _Float16* __restrict__ apack) {
  int t = blockIdx.x * 256 + threadIdx.x;   // 18*2*32 = 1152 records
  if (t >= 18 * 2 * 32) return;
  int lane  = t & 31;
  int mt    = (t >> 5) & 1;
  int chunk = t >> 6;                       // 0..17
  int k     = chunk >> 1;
  int cHalf = chunk & 1;
  int m  = lane & 15;
  int hs = lane >> 4;
  _Float16* rec = apack + (size_t)t * 16;
#pragma unroll
  for (int h = 0; h < 16; ++h) {
    int crel = h + hs * 16;
    int c = cHalf * 32 + crel;
    int o = mt * 16 + m;                    // c_off; zero-pad rows 18..31
    rec[h] = (o < 18) ? (_Float16)wo[((size_t)o * CDIM + c) * 9 + k] : (_Float16)0.0f;
  }
}

// Deform weights: (64, 64, 3, 3); 4 mtiles.
__global__ void __launch_bounds__(256)
pack_wdeform(const float* __restrict__ wd, _Float16* __restrict__ apack) {
  int t = blockIdx.x * 256 + threadIdx.x;   // 18*4*32 = 2304 records
  if (t >= 18 * 4 * 32) return;
  int lane  = t & 31;
  int mt    = (t >> 5) & 3;
  int chunk = t >> 7;
  int k     = chunk >> 1;
  int cHalf = chunk & 1;
  int m  = lane & 15;
  int hs = lane >> 4;
  _Float16* rec = apack + (size_t)t * 16;
#pragma unroll
  for (int h = 0; h < 16; ++h) {
    int crel = h + hs * 16;
    int c = cHalf * 32 + crel;
    int o = mt * 16 + m;
    rec[h] = (_Float16)wd[((size_t)o * CDIM + c) * 9 + k];
  }
}

// ---------------------------------------------------------------------------
// Fused kernel: per 16-pixel tile (one wave):
//   Phase 1: offset conv via WMMA (regular zero-padded NHWC reads) ->
//            dy/dx for the tile's 16 pixels staged in LDS (intra-wave).
//   Phase 2: bilinear 4-corner gather/blend -> f16 B-fragments ->
//            4x v_wmma_f32_16x16x32_f16 per K-chunk across 64 out channels.
// A-fragments for both phases staged in LDS once per block (8-wave reuse).
// ---------------------------------------------------------------------------
__global__ void __launch_bounds__(256)
deform_fused_wmma(const float* __restrict__ xn,
                  const _Float16* __restrict__ aoff_g,
                  const _Float16* __restrict__ adef_g,
                  const float* __restrict__ b_off,
                  const float* __restrict__ b_def,
                  float* __restrict__ out) {
  extern __shared__ char smem[];
  _Float16* aoff_lds = (_Float16*)smem;
  _Float16* adef_lds = (_Float16*)(smem + AOFF_BYTES);
  float*    off_lds  = (float*)(smem + AOFF_BYTES + ADEF_BYTES);

  int t = threadIdx.x;
  // Cooperative stage of both A-fragment packs into LDS (float4 granularity).
  {
    const float4* s0 = (const float4*)aoff_g;
    float4*       d0 = (float4*)aoff_lds;
    for (int i = t; i < AOFF_BYTES / 16; i += 256) d0[i] = s0[i];
    const float4* s1 = (const float4*)adef_g;
    float4*       d1 = (float4*)adef_lds;
    for (int i = t; i < ADEF_BYTES / 16; i += 256) d1[i] = s1[i];
  }
  __syncthreads();

  int lane = t & 31;
  int wave = t >> 5;
  int tile = blockIdx.x * 8 + wave;        // 8192 tiles = B*H*(W/16)
  int pix = lane & 15;
  int hs  = lane >> 4;                     // 0: K rows 0..15, 1: K rows 16..31
  int wt = tile & 7;
  int ho = (tile >> 3) & (HDIM - 1);
  int b  = tile >> 10;
  int wo = wt * 16 + pix;

  const v16h* aoff_v = (const v16h*)aoff_lds;
  const v16h* adef_v = (const v16h*)adef_lds;
  float* myoff = off_lds + (wave * 16 + pix) * 18;

  // ---------------- Phase 1: offset conv (WMMA) ----------------
  {
    v8f aco[2] = {v8f{}, v8f{}};
    for (int k = 0; k < 9; ++k) {
      int ki = k / 3, kj = k % 3;
      int y  = ho - 1 + ki;
      int xx = wo - 1 + kj;
      float flag = (y >= 0 && y < HDIM && xx >= 0 && xx < WDIM) ? 1.0f : 0.0f;
      int cy = min(max(y, 0), HDIM - 1);
      int cx = min(max(xx, 0), WDIM - 1);
      const float* ps = xn + ((size_t)((b * HDIM + cy) * WDIM) + cx) * CDIM;
#pragma unroll
      for (int ch = 0; ch < 2; ++ch) {
        int cbase = ch * 32 + hs * 16;
        v16h bfrag;
#pragma unroll
        for (int i = 0; i < 4; ++i) {
          float4 a = *(const float4*)(ps + cbase + i * 4);
          bfrag[i * 4 + 0] = (_Float16)(flag * a.x);
          bfrag[i * 4 + 1] = (_Float16)(flag * a.y);
          bfrag[i * 4 + 2] = (_Float16)(flag * a.z);
          bfrag[i * 4 + 3] = (_Float16)(flag * a.w);
        }
        int recbase = ((k * 2 + ch) * 2) * 32 + lane;
#pragma unroll
        for (int mt = 0; mt < 2; ++mt) {
          v16h af = aoff_v[recbase + mt * 32];
          aco[mt] = __builtin_amdgcn_wmma_f32_16x16x32_f16(
              false, af, false, bfrag, (short)0, aco[mt], false, false);
        }
      }
    }
    // Redistribute: C/D layout lane n: M = hs*8 + r (+16*mt), N = pix.
#pragma unroll
    for (int mt = 0; mt < 2; ++mt) {
#pragma unroll
      for (int r = 0; r < 8; ++r) {
        int c_off = mt * 16 + hs * 8 + r;
        if (c_off < 18) myoff[c_off] = aco[mt][r] + b_off[c_off];
      }
    }
  }
  // Intra-wave LDS RAW fence (stores above, lane-crossing reads below).
  asm volatile("s_wait_dscnt 0x0" ::: "memory");

  // ---------------- Phase 2: bilinear gather + deform GEMM ----------------
  v8f acc[4] = {v8f{}, v8f{}, v8f{}, v8f{}};
  for (int k = 0; k < 9; ++k) {
    int ki = k / 3, kj = k % 3;
    float dy = myoff[2 * k];
    float dx = myoff[2 * k + 1];
    float py = dy + (float)(ho - 1 + ki);
    float px = dx + (float)(wo - 1 + kj);
    float y0f = floorf(py), x0f = floorf(px);
    float wy1 = py - y0f, wx1 = px - x0f;
    float wy0 = 1.0f - wy1, wx0 = 1.0f - wx1;
    int iy0 = (int)y0f, ix0 = (int)x0f;
    int iy1 = iy0 + 1,  ix1 = ix0 + 1;
    float vy0 = (iy0 >= 0 && iy0 < HDIM) ? 1.0f : 0.0f;
    float vy1 = (iy1 >= 0 && iy1 < HDIM) ? 1.0f : 0.0f;
    float vx0 = (ix0 >= 0 && ix0 < WDIM) ? 1.0f : 0.0f;
    float vx1 = (ix1 >= 0 && ix1 < WDIM) ? 1.0f : 0.0f;
    float w00 = wy0 * wx0 * vy0 * vx0;
    float w01 = wy0 * wx1 * vy0 * vx1;
    float w10 = wy1 * wx0 * vy1 * vx0;
    float w11 = wy1 * wx1 * vy1 * vx1;
    int cy0 = min(max(iy0, 0), HDIM - 1), cy1 = min(max(iy1, 0), HDIM - 1);
    int cx0 = min(max(ix0, 0), WDIM - 1), cx1 = min(max(ix1, 0), WDIM - 1);
    const float* p00 = xn + ((size_t)((b * HDIM + cy0) * WDIM) + cx0) * CDIM;
    const float* p01 = xn + ((size_t)((b * HDIM + cy0) * WDIM) + cx1) * CDIM;
    const float* p10 = xn + ((size_t)((b * HDIM + cy1) * WDIM) + cx0) * CDIM;
    const float* p11 = xn + ((size_t)((b * HDIM + cy1) * WDIM) + cx1) * CDIM;

#pragma unroll
    for (int ch = 0; ch < 2; ++ch) {
      int cbase = ch * 32 + hs * 16;
      float v[16];
#pragma unroll
      for (int i = 0; i < 4; ++i) {
        float4 a0 = *(const float4*)(p00 + cbase + i * 4);
        float4 a1 = *(const float4*)(p01 + cbase + i * 4);
        float4 a2 = *(const float4*)(p10 + cbase + i * 4);
        float4 a3 = *(const float4*)(p11 + cbase + i * 4);
        v[i * 4 + 0] = w00 * a0.x + w01 * a1.x + w10 * a2.x + w11 * a3.x;
        v[i * 4 + 1] = w00 * a0.y + w01 * a1.y + w10 * a2.y + w11 * a3.y;
        v[i * 4 + 2] = w00 * a0.z + w01 * a1.z + w10 * a2.z + w11 * a3.z;
        v[i * 4 + 3] = w00 * a0.w + w01 * a1.w + w10 * a2.w + w11 * a3.w;
      }
      v16h bfrag;
#pragma unroll
      for (int i = 0; i < 16; ++i) bfrag[i] = (_Float16)v[i];

      int recbase = ((k * 2 + ch) * 4) * 32 + lane;
#pragma unroll
      for (int mt = 0; mt < 4; ++mt) {
        v16h afrag = adef_v[recbase + mt * 32];
        acc[mt] = __builtin_amdgcn_wmma_f32_16x16x32_f16(
            false, afrag, false, bfrag, (short)0, acc[mt], false, false);
      }
    }
  }

  // Epilogue: C/D layout -> VGPR r, lanes 0-15: M=r; lanes 16-31: M=r+8.
#pragma unroll
  for (int mt = 0; mt < 4; ++mt) {
#pragma unroll
    for (int r = 0; r < 8; ++r) {
      int o = mt * 16 + hs * 8 + r;
      out[(((size_t)(b * ODIM + o) * HDIM) + ho) * WDIM + wo] = acc[mt][r] + b_def[o];
    }
  }
}

// ---------------------------------------------------------------------------
extern "C" void kernel_launch(void* const* d_in, const int* in_sizes, int n_in,
                              void* d_out, int out_size, void* d_ws, size_t ws_size,
                              hipStream_t stream) {
  (void)in_sizes; (void)n_in; (void)out_size; (void)ws_size;
  const float* x        = (const float*)d_in[0];
  const float* w_offset = (const float*)d_in[1];
  const float* b_offset = (const float*)d_in[2];
  const float* w_deform = (const float*)d_in[3];
  const float* b_deform = (const float*)d_in[4];
  float* out = (float*)d_out;

  char* ws = (char*)d_ws;
  const size_t xn_bytes = (size_t)BDIM * HDIM * WDIM * CDIM * 4;   // 32 MB NHWC
  float*    xn     = (float*)ws;
  _Float16* aoff_g = (_Float16*)(ws + xn_bytes);                   // 36 KB
  _Float16* adef_g = (_Float16*)(ws + xn_bytes + AOFF_BYTES);      // 72 KB

  transpose_nchw_nhwc<<<2048, 256, 0, stream>>>(x, xn);
  pack_woffset<<<5, 256, 0, stream>>>(w_offset, aoff_g);
  pack_wdeform<<<9, 256, 0, stream>>>(w_deform, adef_g);
  deform_fused_wmma<<<(BDIM * HDIM * (WDIM / 16)) / 8, 256, SMEM_BYTES, stream>>>(
      xn, aoff_g, adef_g, b_offset, b_deform, out);
}
